// CompressedMultiHeadAttention_87608742904323
// MI455X (gfx1250) — compile-verified, matrix operand order checked by reading, not run on hardware
//
#include <hip/hip_runtime.h>
#include <hip/hip_bf16.h>

typedef __attribute__((ext_vector_type(16))) _Float16 v16h;
typedef __attribute__((ext_vector_type(8)))  _Float16 v8h;
typedef __attribute__((ext_vector_type(8)))  float    v8f;
typedef __attribute__((ext_vector_type(4)))  unsigned int v4u;
typedef __attribute__((ext_vector_type(8)))  unsigned int v8u;

#define B_  2
#define S_  8192
#define D_  1024
#define H_  32
#define CF_ 8
#define T_  1024   /* S/CF */
#define HD_ 32     /* D/H  */

__device__ __forceinline__ v16h load_frag(const _Float16* p) {
  // A/B 16-bit fragment: two contiguous 8-element (16B) chunks at +0 and +16
  v8h lo = *reinterpret_cast<const v8h*>(p);
  v8h hi = *reinterpret_cast<const v8h*>(p + 16);
  return __builtin_shufflevector(lo, hi, 0,1,2,3,4,5,6,7,8,9,10,11,12,13,14,15);
}

__device__ __forceinline__ v8f wmma16(v16h a, v16h b, v8f c) {
  return __builtin_amdgcn_wmma_f32_16x16x32_f16(false, a, false, b, (short)0, c, false, false);
}

// ---- Tensor Data Mover: async copy of a [rows x 32] f16 tile (row stride = `stride`
// elements) from global into packed LDS (rows*64 bytes). D# per ISA 08_async_tensor §8.
__device__ __forceinline__ void tdm_load_tile(const _Float16* gsrc, unsigned lds_off,
                                              unsigned rows, unsigned stride) {
  unsigned long long ga = (unsigned long long)(size_t)gsrc;
  v4u g0;
  g0[0] = 1u;                                   // count=1, user descriptor
  g0[1] = lds_off;                              // lds_addr (bytes)
  g0[2] = (unsigned)ga;                         // global_addr[31:0]
  g0[3] = (unsigned)(ga >> 32) | (2u << 30);    // global_addr[56:32] | type=2
  v8u g1;
  g1[0] = (1u << 16);                           // workgroup_mask=0, data_size=1 (2B)
  g1[1] = (32u << 16);                          // tensor_dim0[15:0]=32 at bits[63:48]
  g1[2] = (rows << 16);                         // tensor_dim0 hi | tensor_dim1[15:0]
  g1[3] = (32u << 16);                          // tensor_dim1 hi | tile_dim0=32
  g1[4] = rows;                                 // tile_dim1=rows, tile_dim2=0
  g1[5] = stride;                               // tensor_dim0_stride[31:0] (elements)
  g1[6] = 0u;
  g1[7] = 0u;
  asm volatile("tensor_load_to_lds %0, %1" :: "s"(g0), "s"(g1) : "memory");
}

// ---------------- elementwise prep ----------------
__global__ void cvt_f16_kernel(const float* __restrict__ in, _Float16* __restrict__ out, int n) {
  int i = blockIdx.x * blockDim.x + threadIdx.x;
  if (i < n) out[i] = (_Float16)in[i];
}

// Wc [o][i][j] (flat o*8192 + i*8 + j)  ->  Wt [o][j*1024 + i]  (B operand = [N][K])
__global__ void prep_wc_kernel(const float* __restrict__ W, _Float16* __restrict__ Wt, int n) {
  int i = blockIdx.x * blockDim.x + threadIdx.x;
  if (i < n) {
    int o  = i >> 13;
    int r  = i & 8191;
    int ii = r >> 3;
    int j  = r & 7;
    Wt[(size_t)o * 8192 + j * 1024 + ii] = (_Float16)W[i];
  }
}

// ---------------- shared GEMM core: C[256x64] per block, wave tile 32x64 ----------------
// A: per-wave 32x32 slab TDM-staged into private LDS double buffer.
// B: 64x32 slab TDM-staged by wave 0, shared, double buffered.
// Main loop = ds_load_b128 + v_wmma only; all global traffic on the TDM engine.
__device__ __forceinline__ void gemm_core(
    const _Float16* __restrict__ A, const _Float16* __restrict__ Bt, int K,
    int m0, int n0, _Float16* Bsh, _Float16* AshW, int wave, int lane, v8f acc[2][4])
{
  const int ln = lane & 15, hh = lane >> 4, koff = hh * 8;
  const _Float16* abase = A + (size_t)m0 * K;
  const _Float16* bbase = Bt + (size_t)n0 * K;
  const unsigned blds = (unsigned)(size_t)Bsh;     // 2 x 64*32 f16
  const unsigned alds = (unsigned)(size_t)AshW;    // 2 x 32*32 f16 (per wave)

  tdm_load_tile(abase, alds, 32u, (unsigned)K);
  if (wave == 0) tdm_load_tile(bbase, blds, 64u, (unsigned)K);
  __builtin_amdgcn_s_wait_tensorcnt(0);
  __syncthreads();

  int cur = 0;
  for (int kb = 0; kb < K; kb += 32) {
    if (kb + 32 < K) {                               // prefetch next K-slab
      tdm_load_tile(abase + kb + 32, alds + (cur ? 0u : 2048u), 32u, (unsigned)K);
      if (wave == 0)
        tdm_load_tile(bbase + kb + 32, blds + (cur ? 0u : 4096u), 64u, (unsigned)K);
    }
    const _Float16* as = AshW + (cur ? 32 * 32 : 0);
    const _Float16* bs = Bsh + (cur ? 64 * 32 : 0);
    v16h a0 = load_frag(as + ln * 32 + koff);
    v16h a1 = load_frag(as + (16 + ln) * 32 + koff);
#pragma unroll
    for (int t = 0; t < 4; ++t) {
      v16h b = load_frag(bs + (t * 16 + ln) * 32 + koff);
      acc[0][t] = wmma16(a0, b, acc[0][t]);
      acc[1][t] = wmma16(a1, b, acc[1][t]);
    }
    __builtin_amdgcn_s_wait_tensorcnt(0);
    __syncthreads();
    cur ^= 1;
  }
}

// ---------------- fused QKV GEMM: [16384,1024] x [1024,3072] ----------------
__global__ __launch_bounds__(256) void gemm_qkv_kernel(
    const _Float16* __restrict__ A, const _Float16* __restrict__ Bt,
    const float* __restrict__ bq, const float* __restrict__ bkv,
    _Float16* __restrict__ qf, _Float16* __restrict__ kf, _Float16* __restrict__ vf,
    float* __restrict__ kout, float* __restrict__ vout)
{
  __shared__ __align__(16) _Float16 Bsh[2 * 64 * 32];
  __shared__ __align__(16) _Float16 Ash[8][2 * 32 * 32];
  const int wave = threadIdx.x >> 5, lane = threadIdx.x & 31;
  const int ln = lane & 15, hh = lane >> 4;
  const int m0 = blockIdx.y * 256 + wave * 32;
  const int n0 = blockIdx.x * 64;
  v8f acc[2][4] = {{v8f{}, v8f{}, v8f{}, v8f{}}, {v8f{}, v8f{}, v8f{}, v8f{}}};
  gemm_core(A, Bt, D_, m0, n0, Bsh, Ash[wave], wave, lane, acc);
#pragma unroll
  for (int mr = 0; mr < 2; ++mr) {
#pragma unroll
    for (int t = 0; t < 4; ++t) {
      const int n = n0 + t * 16 + ln;
#pragma unroll
      for (int r = 0; r < 8; ++r) {
        const int m = m0 + mr * 16 + r + 8 * hh;
        float v = acc[mr][t][r];
        if (n < 1024) {
          qf[(size_t)m * D_ + n] = (_Float16)(v + bq[n]);
        } else if (n < 2048) {
          const int d = n - 1024;
          v += bkv[d];
          kout[(size_t)m * D_ + d] = v;
          kf[(size_t)m * D_ + d] = (_Float16)v;
        } else {
          const int d = n - 2048;
          v += bkv[d + 1024];
          vout[(size_t)m * D_ + d] = v;
          vf[(size_t)m * D_ + d] = (_Float16)v;
        }
      }
    }
  }
}

// ---------------- compression GEMM: [2048,8192] x [8192,1024] ----------------
__global__ __launch_bounds__(256) void gemm_comp_kernel(
    const _Float16* __restrict__ A, const _Float16* __restrict__ Bt,
    const float* __restrict__ bias, _Float16* __restrict__ outf,
    int K, int transOut)
{
  __shared__ __align__(16) _Float16 Bsh[2 * 64 * 32];
  __shared__ __align__(16) _Float16 Ash[8][2 * 32 * 32];
  const int wave = threadIdx.x >> 5, lane = threadIdx.x & 31;
  const int ln = lane & 15, hh = lane >> 4;
  const int m0 = blockIdx.y * 256 + wave * 32;
  const int n0 = blockIdx.x * 64;
  v8f acc[2][4] = {{v8f{}, v8f{}, v8f{}, v8f{}}, {v8f{}, v8f{}, v8f{}, v8f{}}};
  gemm_core(A, Bt, K, m0, n0, Bsh, Ash[wave], wave, lane, acc);
#pragma unroll
  for (int mr = 0; mr < 2; ++mr) {
#pragma unroll
    for (int t = 0; t < 4; ++t) {
      const int n = n0 + t * 16 + ln;
#pragma unroll
      for (int r = 0; r < 8; ++r) {
        const int m = m0 + mr * 16 + r + 8 * hh;
        const float v = acc[mr][t][r] + bias[n];
        if (!transOut) {
          outf[(size_t)m * D_ + n] = (_Float16)v;                  // kc: [b][t][d]
        } else {
          const int bb = m >> 10, tt = m & 1023;
          outf[((size_t)(bb * D_ + n)) * T_ + tt] = (_Float16)v;   // vt: [b][d][t]
        }
      }
    }
  }
}

// ---------------- flash attention over compressed K/V ----------------
// Block = 8 waves x 16 queries of one (b,h). K/V chunks (32 kv-positions) are
// TDM-staged into LDS once per block (shared by all waves), double buffered.
// Chunk loop is block-uniform; causal masking makes over-run chunks exact no-ops.
__global__ __launch_bounds__(256) void attn_kernel(
    const _Float16* __restrict__ qf, const _Float16* __restrict__ kcf,
    const _Float16* __restrict__ vtf, float* __restrict__ out)
{
  __shared__ __align__(16) _Float16 Ksh[2 * 32 * 32];   // [t_local][d]
  __shared__ __align__(16) _Float16 Vsh[2 * 32 * 32];   // [d_local][t]  (V^T)
  __shared__ __align__(16) _Float16 Pbuf[8][16 * 32];
  const int wave = threadIdx.x >> 5, lane = threadIdx.x & 31;
  const int ln = lane & 15, hh = lane >> 4, koff = hh * 8;
  const int b = blockIdx.y >> 5;
  const int h = blockIdx.y & 31;
  const int qblock = blockIdx.x * 128;
  const int qbase = qblock + wave * 16;
  const float SCALE = 0.17677669529663687f;   // 1/sqrt(32)
  const float NEGINF = -__builtin_inff();

  v16h aq = load_frag(qf + ((size_t)(b * S_ + qbase + ln)) * D_ + h * HD_ + koff);

  float mi[8], li[8];
  v8f o0 = v8f{}, o1 = v8f{};
#pragma unroll
  for (int r = 0; r < 8; ++r) { mi[r] = NEGINF; li[r] = 0.f; }

  const _Float16* kb_p = kcf + (size_t)b * T_ * D_ + h * HD_;   // rows t, stride D
  const _Float16* vb_p = vtf + (size_t)b * D_ * T_ + (size_t)h * HD_ * T_; // rows d, stride T
  _Float16* P = Pbuf[wave];
  const unsigned klds = (unsigned)(size_t)Ksh;
  const unsigned vlds = (unsigned)(size_t)Vsh;

  int ncols = qblock + 128; if (ncols > T_) ncols = T_;   // block-uniform causal bound
  const int nch = (ncols + 31) >> 5;

  if (wave == 0) {
    tdm_load_tile(kb_p, klds, 32u, D_);
    tdm_load_tile(vb_p, vlds, 32u, T_);
  }
  __builtin_amdgcn_s_wait_tensorcnt(0);
  __syncthreads();

  int cur = 0;
  for (int ch = 0; ch < nch; ++ch) {
    const int c0 = ch * 32;
    if (wave == 0 && ch + 1 < nch) {                      // prefetch next chunk
      tdm_load_tile(kb_p + (size_t)(c0 + 32) * D_, klds + (cur ? 0u : 2048u), 32u, D_);
      tdm_load_tile(vb_p + (c0 + 32),              vlds + (cur ? 0u : 2048u), 32u, T_);
    }
    const _Float16* ks = Ksh + (cur ? 32 * 32 : 0);
    const _Float16* vs = Vsh + (cur ? 32 * 32 : 0);
    // ---- S = q . k^T (two 16-col tiles) ----
    v16h bk0 = load_frag(ks + (ln)      * 32 + koff);
    v16h bk1 = load_frag(ks + (16 + ln) * 32 + koff);
    v8f s0 = wmma16(aq, bk0, v8f{});
    v8f s1 = wmma16(aq, bk1, v8f{});
    const int col0 = c0 + ln, col1 = col0 + 16;
#pragma unroll
    for (int r = 0; r < 8; ++r) {
      const int qi = qbase + r + 8 * hh;
      float x0 = (col0 <= qi) ? s0[r] * SCALE : NEGINF;
      float x1 = (col1 <= qi) ? s1[r] * SCALE : NEGINF;
      float t = fmaxf(x0, x1);
      t = fmaxf(t, __shfl_xor(t, 1, 32));
      t = fmaxf(t, __shfl_xor(t, 2, 32));
      t = fmaxf(t, __shfl_xor(t, 4, 32));
      t = fmaxf(t, __shfl_xor(t, 8, 32));
      const float mn = fmaxf(mi[r], t);            // stays finite; fully-masked chunk => no-op
      const float alpha = __expf(mi[r] - mn);
      const float p0 = __expf(x0 - mn);
      const float p1 = __expf(x1 - mn);
      float rs = p0 + p1;
      rs += __shfl_xor(rs, 1, 32);
      rs += __shfl_xor(rs, 2, 32);
      rs += __shfl_xor(rs, 4, 32);
      rs += __shfl_xor(rs, 8, 32);
      li[r] = li[r] * alpha + rs;
      mi[r] = mn;
      o0[r] *= alpha;
      o1[r] *= alpha;
      const int prow = r + 8 * hh;
      P[prow * 32 + ln] = (_Float16)p0;
      P[prow * 32 + 16 + ln] = (_Float16)p1;
    }
    asm volatile("s_wait_dscnt 0" ::: "memory");   // intra-wave cross-lane LDS handoff
    v16h ap  = load_frag(P + ln * 32 + koff);
    v16h bv0 = load_frag(vs + (ln)      * 32 + koff);
    v16h bv1 = load_frag(vs + (16 + ln) * 32 + koff);
    o0 = wmma16(ap, bv0, o0);
    o1 = wmma16(ap, bv1, o1);
    __builtin_amdgcn_s_wait_tensorcnt(0);
    __syncthreads();
    cur ^= 1;
  }
#pragma unroll
  for (int r = 0; r < 8; ++r) {
    const int qi = qbase + r + 8 * hh;
    const float inv = 1.0f / li[r];
    float* op = out + ((size_t)(b * S_ + qi)) * D_ + h * HD_;
    op[ln] = o0[r] * inv;
    op[16 + ln] = o1[r] * inv;
  }
}

extern "C" void kernel_launch(void* const* d_in, const int* in_sizes, int n_in,
                              void* d_out, int out_size, void* d_ws, size_t ws_size,
                              hipStream_t stream) {
  (void)in_sizes; (void)n_in; (void)out_size; (void)ws_size;
  const float* x   = (const float*)d_in[0];
  const float* Wq  = (const float*)d_in[1];
  const float* bq  = (const float*)d_in[2];
  const float* Wkv = (const float*)d_in[3];
  const float* bkv = (const float*)d_in[4];
  const float* Wck = (const float*)d_in[5];
  const float* bck = (const float*)d_in[6];
  const float* Wcv = (const float*)d_in[7];
  const float* bcv = (const float*)d_in[8];

  const size_t BSD = (size_t)B_ * S_ * D_;
  float* outp = (float*)d_out;
  float* kout = outp + BSD;
  float* vout = kout + BSD;

  _Float16* ws = (_Float16*)d_ws;
  size_t off = 0;
  _Float16* qf   = ws + off; off += BSD;
  _Float16* kf   = ws + off; off += BSD;
  _Float16* vf   = ws + off; off += BSD;
  _Float16* xf   = ws + off; off += BSD;
  _Float16* wqkv = ws + off; off += (size_t)3 * D_ * D_;
  _Float16* wckt = ws + off; off += (size_t)CF_ * D_ * D_;
  _Float16* wcvt = ws + off; off += (size_t)CF_ * D_ * D_;
  _Float16* kcf  = ws + off; off += (size_t)B_ * T_ * D_;
  _Float16* vtf  = ws + off; off += (size_t)B_ * T_ * D_;

  const int nx = B_ * S_ * D_;
  cvt_f16_kernel<<<(nx + 255) / 256, 256, 0, stream>>>(x, xf, nx);
  cvt_f16_kernel<<<(D_ * D_ + 255) / 256, 256, 0, stream>>>(Wq, wqkv, D_ * D_);
  cvt_f16_kernel<<<(2 * D_ * D_ + 255) / 256, 256, 0, stream>>>(Wkv, wqkv + D_ * D_, 2 * D_ * D_);
  prep_wc_kernel<<<(CF_ * D_ * D_ + 255) / 256, 256, 0, stream>>>(Wck, wckt, CF_ * D_ * D_);
  prep_wc_kernel<<<(CF_ * D_ * D_ + 255) / 256, 256, 0, stream>>>(Wcv, wcvt, CF_ * D_ * D_);

  gemm_qkv_kernel<<<dim3(3072 / 64, 16384 / 256), 256, 0, stream>>>(
      xf, wqkv, bq, bkv, qf, kf, vf, kout, vout);

  gemm_comp_kernel<<<dim3(1024 / 64, 2048 / 256), 256, 0, stream>>>(
      kf, wckt, bck, kcf, CF_ * D_, 0);
  gemm_comp_kernel<<<dim3(1024 / 64, 2048 / 256), 256, 0, stream>>>(
      vf, wcvt, bcv, vtf, CF_ * D_, 1);

  attn_kernel<<<dim3(S_ / 128, B_ * H_), 256, 0, stream>>>(qf, kcf, vtf, outp);
}
